// AdaptiveSampler_51908974739496
// MI455X (gfx1250) — matrix operand order repaired and checked by reading, not compiled, not run on hardware
//
#include <hip/hip_runtime.h>
#include <hip/hip_bf16.h>

typedef __attribute__((ext_vector_type(16))) _Float16 v16h;
typedef __attribute__((ext_vector_type(8)))  _Float16 v8h;
typedef __attribute__((ext_vector_type(8)))  float    v8f;

#define B_   32
#define C_   1024
#define H_   64
#define W_   64
#define HW_  (H_*W_)
#define J_   128
#define NP_  4
#define HID_ 128
#define P_   (B_*J_)    // 4096 points
#define OC_  256        // concatenated hidden rows (128 off | 128 att)
#define CC_  32         // channels per gather block

// ---------------------------------------------------------------------------
// Async staging helpers (gfx1250): stream one 16KB channel plane into LDS via
// global_load_async_to_lds_b128 (each lane moves 16B; 4 issues x 256 thr).
// Tracked by ASYNCcnt; consumed after s_wait_asynccnt 0 + barrier.
// ---------------------------------------------------------------------------
__device__ __forceinline__ void async_plane_load(uint32_t lds_base,
                                                 const float* __restrict__ gsrc,
                                                 int t) {
#pragma unroll
    for (int q = 0; q < 4; ++q) {
        uint32_t loff = lds_base + (uint32_t)(t * 16 + q * 4096);
        uint64_t ga   = (uint64_t)(uintptr_t)(gsrc + t * 4 + q * 1024);
        asm volatile("global_load_async_to_lds_b128 %0, %1, off"
                     :: "v"(loff), "v"(ga) : "memory");
    }
}
__device__ __forceinline__ void wait_async() {
    asm volatile("s_wait_asynccnt 0x0" ::: "memory");
}
__device__ __forceinline__ uint32_t lds_off_of(const void* p) {
    return (uint32_t)(uintptr_t)p;   // generic LDS addr = {aperture_hi, offset}
}

// ---------------------------------------------------------------------------
// 1) Convert concat(w_off1, w_att1) [256 x 1024] f32 -> f16 row-major (WMMA A)
// ---------------------------------------------------------------------------
__global__ __launch_bounds__(256) void k_convert_w(const float* __restrict__ w_off1,
                                                   const float* __restrict__ w_att1,
                                                   _Float16* __restrict__ w1h) {
    int i = blockIdx.x * 256 + threadIdx.x;      // 0 .. 256*1024-1
    int r = i >> 10;
    int c = i & (C_ - 1);
    float v = (r < HID_) ? w_off1[r * C_ + c] : w_att1[(r - HID_) * C_ + c];
    w1h[i] = (_Float16)v;
}

// ---------------------------------------------------------------------------
// 2) Seed gather: async-stage each channel plane into LDS (double-buffered),
//    then bilinear-gather all 128 keypoints from LDS. seed stored [P][C] f16
//    so the GEMM B-fragment reads contiguous K.
// ---------------------------------------------------------------------------
__global__ __launch_bounds__(256) void k_seed(const float* __restrict__ feat,
                                              const float* __restrict__ kp,
                                              _Float16* __restrict__ seedh) {
    __shared__ float sbuf[2][HW_];               // 2 x 16KB plane buffers
    __shared__ int   sidx[J_][4];
    __shared__ float swgt[J_][4];
    int b  = blockIdx.x / (C_ / CC_);
    int cb = (blockIdx.x % (C_ / CC_)) * CC_;
    int t  = threadIdx.x;
    const float* fb = feat + (size_t)b * C_ * HW_;
    uint32_t lb0 = lds_off_of(&sbuf[0][0]);
    uint32_t lb1 = lds_off_of(&sbuf[1][0]);

    async_plane_load(lb0, fb + (size_t)cb * HW_, t);   // prefetch first plane

    if (t < J_) {
        int p = b * J_ + t;
        float gx = kp[p * 2 + 0], gy = kp[p * 2 + 1];
        float ix = (gx + 1.f) * 0.5f * (W_ - 1);
        float iy = (gy + 1.f) * 0.5f * (H_ - 1);
        float x0f = floorf(ix), y0f = floorf(iy);
        float wx1 = ix - x0f, wx0 = 1.f - wx1;
        float wy1 = iy - y0f, wy0 = 1.f - wy1;
        float xs[2]  = {x0f, x0f + 1.f};
        float ys[2]  = {y0f, y0f + 1.f};
        float wxs[2] = {wx0, wx1};
        float wys[2] = {wy0, wy1};
#pragma unroll
        for (int k = 0; k < 4; ++k) {
            float xf = xs[k & 1], yf = ys[k >> 1];
            bool valid = (xf >= 0.f) && (xf <= (float)(W_ - 1)) &&
                         (yf >= 0.f) && (yf <= (float)(H_ - 1));
            int xi = (int)fminf(fmaxf(xf, 0.f), (float)(W_ - 1));
            int yi = (int)fminf(fmaxf(yf, 0.f), (float)(H_ - 1));
            sidx[t][k] = yi * W_ + xi;
            swgt[t][k] = valid ? wxs[k & 1] * wys[k >> 1] : 0.f;
        }
    }
    wait_async();
    __syncthreads();

    int j = t >> 1, half = t & 1;                // thread pair splits 4 corners
    for (int i = 0; i < CC_; ++i) {
        int cur = i & 1;
        if (i + 1 < CC_)                          // prefetch next plane
            async_plane_load(cur ? lb0 : lb1, fb + (size_t)(cb + i + 1) * HW_, t);
        const float* pl = sbuf[cur];
        int k0 = 2 * half;
        float acc = swgt[j][k0 + 0] * pl[sidx[j][k0 + 0]]
                  + swgt[j][k0 + 1] * pl[sidx[j][k0 + 1]];
        acc += __shfl_xor(acc, 1, 32);            // combine corner halves
        if (half == 0)
            seedh[(size_t)(b * J_ + j) * C_ + (cb + i)] = (_Float16)acc;
        wait_async();
        __syncthreads();
    }
}

// ---------------------------------------------------------------------------
// 3) Fused layer-1 GEMM: h[256][4096] = relu(W1cat(f16) x seed(f16) + b1cat)
//    Wave-per-tile WMMA, K = 1024 in steps of 32 (contiguous k = 16h + s in A and B).
// ---------------------------------------------------------------------------
__global__ __launch_bounds__(256) void k_gemm1(const _Float16* __restrict__ w1h,
                                               const _Float16* __restrict__ seedh,
                                               const float* __restrict__ b_off1,
                                               const float* __restrict__ b_att1,
                                               float* __restrict__ hbuf) {
    int lane = threadIdx.x & 31;
    int wid  = threadIdx.x >> 5;
    int tile = blockIdx.x * 8 + wid;              // 0..4095
    int tm = tile >> 8;                           // 0..15  (256 rows / 16)
    int tn = tile & 255;                          // 0..255 (4096 cols / 16)

    int mrow = tm * 16 + (lane & 15);
    int ksel = lane >> 4;
    const v8h* a0 = (const v8h*)(w1h + (size_t)mrow * C_ + ksel * 16);
    const v8h* a1 = (const v8h*)(w1h + (size_t)mrow * C_ + 8 + ksel * 16);

    int pcol = tn * 16 + (lane & 15);
    const _Float16* bp = seedh + (size_t)pcol * C_ + ksel * 16;

    v8f acc = {};
    for (int kb = 0; kb < C_ / 32; ++kb) {
        v8h r0 = a0[kb * 4];
        v8h r1 = a1[kb * 4];
        v16h afrag;
#pragma unroll
        for (int e = 0; e < 8; ++e) { afrag[e] = r0[e]; afrag[8 + e] = r1[e]; }
        v16h bfrag = *(const v16h*)(bp + kb * 32);
        acc = __builtin_amdgcn_wmma_f32_16x16x32_f16(
            false, afrag, false, bfrag, (short)0, acc, false, false);
    }

    int n     = tn * 16 + (lane & 15);
    int mbase = tm * 16 + (lane >> 4) * 8;
#pragma unroll
    for (int e = 0; e < 8; ++e) {
        int r = mbase + e;
        float bias = (r < HID_) ? b_off1[r] : b_att1[r - HID_];
        float v = acc[e] + bias;
        hbuf[(size_t)r * P_ + n] = v > 0.f ? v : 0.f;
    }
}

// ---------------------------------------------------------------------------
// 4) Heads: per point, 256->12 dense layer, then new grid + softmax weights.
// ---------------------------------------------------------------------------
__global__ __launch_bounds__(128) void k_head(const float* __restrict__ hbuf,
                                              const float* __restrict__ kp,
                                              const float* __restrict__ w_off2,
                                              const float* __restrict__ b_off2,
                                              const float* __restrict__ w_att2,
                                              const float* __restrict__ b_att2,
                                              float* __restrict__ grid2,
                                              float* __restrict__ attw) {
    __shared__ float ho[HID_], ha[HID_], res[12];
    int p = blockIdx.x;
    int t = threadIdx.x;
    ho[t] = hbuf[(size_t)t * P_ + p];
    ha[t] = hbuf[(size_t)(HID_ + t) * P_ + p];
    __syncthreads();
    if (t < 12) {
        float acc;
        if (t < 8) {
            acc = b_off2[t];
            for (int i = 0; i < HID_; ++i) acc += w_off2[t * HID_ + i] * ho[i];
        } else {
            int o = t - 8;
            acc = b_att2[o];
            for (int i = 0; i < HID_; ++i) acc += w_att2[o * HID_ + i] * ha[i];
        }
        res[t] = acc;
    }
    __syncthreads();
    if (t == 0) {
        float gx = kp[p * 2 + 0], gy = kp[p * 2 + 1];
        const float sx = 2.f / (W_ - 1), sy = 2.f / (H_ - 1);
        float l0 = res[8], l1 = res[9], l2 = res[10], l3 = res[11];
        float m  = fmaxf(fmaxf(l0, l1), fmaxf(l2, l3));
        float e0 = __expf(l0 - m), e1 = __expf(l1 - m);
        float e2 = __expf(l2 - m), e3 = __expf(l3 - m);
        float inv = 1.f / (e0 + e1 + e2 + e3);
        float wsft[4] = {e0 * inv, e1 * inv, e2 * inv, e3 * inv};
#pragma unroll
        for (int n = 0; n < NP_; ++n) {
            grid2[(p * NP_ + n) * 2 + 0] = gx + res[2 * n + 0] * sx;
            grid2[(p * NP_ + n) * 2 + 1] = gy + res[2 * n + 1] * sy;
            attw[p * NP_ + n] = wsft[n];
        }
    }
}

// ---------------------------------------------------------------------------
// 5) Final fused gather: out[b][j][c] = sum_n attw * bilinear(feat, grid2).
//    Async-stage channel planes into LDS (double-buffered), gather from LDS.
//    Attention + zero-pad folded into per-corner weights.
// ---------------------------------------------------------------------------
__global__ __launch_bounds__(256) void k_fuse(const float* __restrict__ feat,
                                              const float* __restrict__ grid2,
                                              const float* __restrict__ attw,
                                              float* __restrict__ out) {
    __shared__ float sbuf[2][HW_];               // 32 KB plane buffers
    __shared__ int   cidx[J_ * NP_][4];          // 8 KB
    __shared__ float cwgt[J_ * NP_][4];          // 8 KB
    int b  = blockIdx.x / (C_ / CC_);
    int cb = (blockIdx.x % (C_ / CC_)) * CC_;
    int t  = threadIdx.x;
    const float* fb = feat + (size_t)b * C_ * HW_;
    uint32_t lb0 = lds_off_of(&sbuf[0][0]);
    uint32_t lb1 = lds_off_of(&sbuf[1][0]);

    async_plane_load(lb0, fb + (size_t)cb * HW_, t);   // prefetch first plane

    for (int s = t; s < J_ * NP_; s += 256) {
        int j = s >> 2, n = s & 3;
        int p = b * J_ + j;
        float gx = grid2[(p * NP_ + n) * 2 + 0];
        float gy = grid2[(p * NP_ + n) * 2 + 1];
        float aw = attw[p * NP_ + n];
        float ix = (gx + 1.f) * 0.5f * (W_ - 1);
        float iy = (gy + 1.f) * 0.5f * (H_ - 1);
        float x0f = floorf(ix), y0f = floorf(iy);
        float wx1 = ix - x0f, wx0 = 1.f - wx1;
        float wy1 = iy - y0f, wy0 = 1.f - wy1;
        float xs[2]  = {x0f, x0f + 1.f};
        float ys[2]  = {y0f, y0f + 1.f};
        float wxs[2] = {wx0, wx1};
        float wys[2] = {wy0, wy1};
#pragma unroll
        for (int k = 0; k < 4; ++k) {
            float xf = xs[k & 1], yf = ys[k >> 1];
            bool valid = (xf >= 0.f) && (xf <= (float)(W_ - 1)) &&
                         (yf >= 0.f) && (yf <= (float)(H_ - 1));
            int xi = (int)fminf(fmaxf(xf, 0.f), (float)(W_ - 1));
            int yi = (int)fminf(fmaxf(yf, 0.f), (float)(H_ - 1));
            cidx[s][k] = yi * W_ + xi;
            cwgt[s][k] = valid ? aw * wxs[k & 1] * wys[k >> 1] : 0.f;
        }
    }
    wait_async();
    __syncthreads();

    int j = t >> 1, half = t & 1;                // thread pair splits n in halves
    for (int i = 0; i < CC_; ++i) {
        int cur = i & 1;
        if (i + 1 < CC_)
            async_plane_load(cur ? lb0 : lb1, fb + (size_t)(cb + i + 1) * HW_, t);
        const float* pl = sbuf[cur];
        float acc = 0.f;
#pragma unroll
        for (int nn = 0; nn < 2; ++nn) {         // this half's 2 sample points
            int s = j * NP_ + 2 * half + nn;
            acc += cwgt[s][0] * pl[cidx[s][0]];
            acc += cwgt[s][1] * pl[cidx[s][1]];
            acc += cwgt[s][2] * pl[cidx[s][2]];
            acc += cwgt[s][3] * pl[cidx[s][3]];
        }
        acc += __shfl_xor(acc, 1, 32);           // combine the two halves
        if (half == 0)
            out[(size_t)(b * J_ + j) * C_ + (cb + i)] = acc;
        wait_async();
        __syncthreads();
    }
}

// ---------------------------------------------------------------------------
extern "C" void kernel_launch(void* const* d_in, const int* in_sizes, int n_in,
                              void* d_out, int out_size, void* d_ws, size_t ws_size,
                              hipStream_t stream) {
    const float* feat   = (const float*)d_in[0];  // [32,1024,64,64]
    const float* kp     = (const float*)d_in[1];  // [32,128,2]
    const float* w_off1 = (const float*)d_in[2];  // [128,1024]
    const float* b_off1 = (const float*)d_in[3];  // [128]
    const float* w_off2 = (const float*)d_in[4];  // [8,128]
    const float* b_off2 = (const float*)d_in[5];  // [8]
    const float* w_att1 = (const float*)d_in[6];  // [128,1024]
    const float* b_att1 = (const float*)d_in[7];  // [128]
    const float* w_att2 = (const float*)d_in[8];  // [4,128]
    const float* b_att2 = (const float*)d_in[9];  // [4]
    float* out = (float*)d_out;                   // [32,128,1024]

    char* ws = (char*)d_ws;
    _Float16* w1h   = (_Float16*)(ws);                                   // 512 KB
    _Float16* seedh = (_Float16*)(ws + (512u << 10));                    // 8 MB
    float*    hbuf  = (float*)   (ws + (512u << 10) + (8u << 20));       // 4 MB
    float*    g2    = (float*)   (ws + (512u << 10) + (12u << 20));      // 128 KB
    float*    aw    = (float*)   (ws + (512u << 10) + (12u << 20) + (128u << 10)); // 64 KB

    k_convert_w<<<(OC_ * C_) / 256, 256, 0, stream>>>(w_off1, w_att1, w1h);
    k_seed<<<B_ * (C_ / CC_), 256, 0, stream>>>(feat, kp, seedh);
    k_gemm1<<<(16 * 256) / 8, 256, 0, stream>>>(w1h, seedh, b_off1, b_att1, hbuf);
    k_head<<<P_, 128, 0, stream>>>(hbuf, kp, w_off2, b_off2, w_att2, b_att2, g2, aw);
    k_fuse<<<B_ * (C_ / CC_), 256, 0, stream>>>(feat, g2, aw, out);
}
